// SOMLayer_59949153517766
// MI455X (gfx1250) — matrix operand order repaired and checked by reading, compiled, not run on hardware
//
#include <hip/hip_runtime.h>

typedef __attribute__((ext_vector_type(16))) __bf16 v16bf;
typedef __attribute__((ext_vector_type(8)))  __bf16 v8bf;
typedef __attribute__((ext_vector_type(8)))  float  v8f;

#define D_DIM   512
#define K_NODES 1024
#define T_SEQ   1024
#define B_BATCH 64
#define ROWS    32          // z rows per workgroup
#define THREADS 512         // 16 waves
#define KSTEPS  (D_DIM / 32)

static __device__ __forceinline__ v16bf cat8(v8bf a, v8bf b) {
  return __builtin_shufflevector(a, b, 0, 1, 2, 3, 4, 5, 6, 7,
                                       8, 9, 10, 11, 12, 13, 14, 15);
}

// ---------------------------------------------------------------------------
// Kernel 1: nodes -> bf16 hi/lo split (L2-resident, 2 MB) + ||n||^2 per node.
// ---------------------------------------------------------------------------
__global__ __launch_bounds__(256) void som_prep_nodes(
    const float* __restrict__ nodes,
    __bf16* __restrict__ nh, __bf16* __restrict__ nl,
    float* __restrict__ nsq) {
  const int k = blockIdx.x;       // node index 0..1023
  const int tid = threadIdx.x;    // 256 threads
  const float* row = nodes + (size_t)k * D_DIM;
  float s = 0.f;
  for (int i = tid; i < D_DIM; i += 256) {
    float v = row[i];
    __bf16 h = (__bf16)v;
    nh[(size_t)k * D_DIM + i] = h;
    nl[(size_t)k * D_DIM + i] = (__bf16)(v - (float)h);
    s += v * v;
  }
#pragma unroll
  for (int o = 16; o > 0; o >>= 1) s += __shfl_xor(s, o, 32);
  __shared__ float wsum[8];
  if ((tid & 31) == 0) wsum[tid >> 5] = s;
  __syncthreads();
  if (tid == 0) {
    float t = 0.f;
#pragma unroll
    for (int i = 0; i < 8; ++i) t += wsum[i];
    nsq[k] = t;
  }
}

// ---------------------------------------------------------------------------
// Kernel 2: fused split-bf16 WMMA GEMM + SOM epilogue.
// Grid: (B*T)/ROWS workgroups of 512 threads (16 waves).
// Wave mapping: stripe = wave>>3 (16-row stripe), wcol = wave&7 (128 cols).
// ---------------------------------------------------------------------------
__global__ __launch_bounds__(THREADS) void som_main(
    const float* __restrict__ z, const float* __restrict__ nodes,
    const __bf16* __restrict__ nh, const __bf16* __restrict__ nl,
    const float* __restrict__ nsq,
    float* __restrict__ out_som, float* __restrict__ out_q,
    int* __restrict__ out_bmu) {
  extern __shared__ __bf16 smem[];          // 64 KB dynamic LDS
  __bf16* Ah = smem;                        // [ROWS][D_DIM] weighted-z hi
  __bf16* Al = smem + ROWS * D_DIM;         // [ROWS][D_DIM] weighted-z lo
  __shared__ float zsq[ROWS];
  __shared__ float rs[2][8][16];            // per-(stripe,wave) row-sum parts
  __shared__ unsigned long long mkp[2][8][16];
  __shared__ float rsumF[ROWS];
  __shared__ unsigned long long mkF[ROWS];

  const int tid = threadIdx.x;
  const long long rowbase = (long long)blockIdx.x * ROWS;

  // ---- Phase 1: load z rows, apply time decay, split to bf16 hi/lo, |wz|^2.
  {
    const int r  = tid >> 4;                 // row 0..31
    const int c0 = (tid & 15) * 32;          // 32 contiguous floats
    const long long grow = rowbase + r;
    const int t = (int)(grow & (T_SEQ - 1)); // row = b*T + t
    // decay computed in double then cast to f32 (matches reference recipe)
    const float w = (float)pow(0.9, (double)(T_SEQ - 1 - t));
    const float* zr = z + grow * (long long)D_DIM + c0;
    float part = 0.f;
#pragma unroll
    for (int i = 0; i < 32; i += 4) {
      float4 v4 = *(const float4*)(zr + i);
      float vv[4] = {v4.x * w, v4.y * w, v4.z * w, v4.w * w};
#pragma unroll
      for (int u = 0; u < 4; ++u) {
        float v = vv[u];
        __bf16 h = (__bf16)v;
        Ah[r * D_DIM + c0 + i + u] = h;
        Al[r * D_DIM + c0 + i + u] = (__bf16)(v - (float)h);
        part += v * v;
      }
    }
#pragma unroll
    for (int o = 8; o > 0; o >>= 1) part += __shfl_xor(part, o, 32);
    if ((tid & 15) == 0) zsq[r] = part;
  }
  __syncthreads();

  // ---- Phase 2: cross = wz . nodes^T via split-bf16 WMMA (hi*hi+hi*lo+lo*hi)
  const int wave = tid >> 5, lane = tid & 31;
  const int stripe = wave >> 3, wcol = wave & 7;
  const int half = lane >> 4, lm = lane & 15;

  v8f acc[8];
#pragma unroll
  for (int j = 0; j < 8; ++j) acc[j] = (v8f){0.f, 0.f, 0.f, 0.f, 0.f, 0.f, 0.f, 0.f};

  const __bf16* Abase_h = Ah + (stripe * 16 + lm) * D_DIM;
  const __bf16* Abase_l = Al + (stripe * 16 + lm) * D_DIM;
  const int a_off0 = half * 16;       // A frag: 16 contiguous K at h*16
  const int a_off1 = half * 16 + 8;
  const int b_off  = half * 16;       // B frag: 16 contiguous K at h*16

  for (int ks = 0; ks < KSTEPS; ++ks) {
    const int k0 = ks * 32;
    v16bf ah = cat8(*(const v8bf*)(Abase_h + k0 + a_off0),
                    *(const v8bf*)(Abase_h + k0 + a_off1));
    v16bf al = cat8(*(const v8bf*)(Abase_l + k0 + a_off0),
                    *(const v8bf*)(Abase_l + k0 + a_off1));
#pragma unroll
    for (int j = 0; j < 8; ++j) {
      const int ncol = ((wcol << 3) + j) * 16 + lm;   // B-col = node index
      v16bf bh = *(const v16bf*)(nh + (size_t)ncol * D_DIM + k0 + b_off);
      v16bf bl = *(const v16bf*)(nl + (size_t)ncol * D_DIM + k0 + b_off);
      acc[j] = __builtin_amdgcn_wmma_f32_16x16x32_bf16(
          false, ah, false, bh, (short)0, acc[j], false, false);
      acc[j] = __builtin_amdgcn_wmma_f32_16x16x32_bf16(
          false, ah, false, bl, (short)0, acc[j], false, false);
      acc[j] = __builtin_amdgcn_wmma_f32_16x16x32_bf16(
          false, al, false, bh, (short)0, acc[j], false, false);
    }
  }

  // ---- Phase 3: dist / q / argmin epilogue in registers (qu replaces cross).
  float rowq[8];
  unsigned long long mk[8];
#pragma unroll
  for (int v = 0; v < 8; ++v) { rowq[v] = 0.f; mk[v] = ~0ULL; }

#pragma unroll
  for (int j = 0; j < 8; ++j) {
    const int ncol = ((wcol << 3) + j) * 16 + lm;
    const float nsqv = nsq[ncol];
#pragma unroll
    for (int v = 0; v < 8; ++v) {
      const int m = stripe * 16 + half * 8 + v;   // C layout: M = v + 8*half
      float d2 = zsq[m] + nsqv - 2.f * acc[j][v];
      float dist = sqrtf(fmaxf(d2, 0.f));
      float qu = 1.f / (1.f + dist);              // ALPHA = 1
      acc[j][v] = qu;
      rowq[v] += qu;
      unsigned long long key =
          ((unsigned long long)__float_as_uint(dist) << 32) | (unsigned int)ncol;
      mk[v] = (key < mk[v]) ? key : mk[v];        // ties -> smallest index
    }
  }
  // deterministic reduce across the 16 column-lanes of each half
#pragma unroll
  for (int v = 0; v < 8; ++v) {
#pragma unroll
    for (int o = 8; o > 0; o >>= 1) {
      rowq[v] += __shfl_xor(rowq[v], o, 32);
      unsigned long long other = __shfl_xor(mk[v], o, 32);
      mk[v] = (other < mk[v]) ? other : mk[v];
    }
  }
  if (lm == 0) {
#pragma unroll
    for (int v = 0; v < 8; ++v) {
      rs[stripe][wcol][half * 8 + v]  = rowq[v];
      mkp[stripe][wcol][half * 8 + v] = mk[v];
    }
  }
  __syncthreads();

  if (tid < ROWS) {            // final 8-wave combine, deterministic order
    const int s2 = tid >> 4, rr = tid & 15;
    float t = 0.f;
    unsigned long long mv = ~0ULL;
#pragma unroll
    for (int w = 0; w < 8; ++w) {
      t += rs[s2][w][rr];
      unsigned long long c = mkp[s2][w][rr];
      mv = (c < mv) ? c : mv;
    }
    rsumF[tid] = t;
    mkF[tid] = mv;
    out_bmu[rowbase + tid] = (int)(unsigned int)(mv & 0xffffffffULL);
  }
  __syncthreads();

  // ---- q writeout (L1-normalized)
  {
    float denom[8];
#pragma unroll
    for (int v = 0; v < 8; ++v) denom[v] = rsumF[stripe * 16 + half * 8 + v];
#pragma unroll
    for (int j = 0; j < 8; ++j) {
      const int ncol = ((wcol << 3) + j) * 16 + lm;
#pragma unroll
      for (int v = 0; v < 8; ++v) {
        const long long grow = rowbase + stripe * 16 + half * 8 + v;
        out_q[grow * K_NODES + ncol] = acc[j][v] / denom[v];
      }
    }
  }

  // ---- som_z = z + 0.1*(nodes[bmu] - z), float4 vectorized
  for (int i = tid; i < ROWS * (D_DIM / 4); i += THREADS) {
    const int r = i >> 7;                 // D_DIM/4 == 128
    const int c4 = (i & 127) << 2;
    const long long grow = rowbase + r;
    const int bmu = (int)(unsigned int)(mkF[r] & 0xffffffffULL);
    const float4 zv = *(const float4*)(z + grow * (long long)D_DIM + c4);
    const float4 nv = *(const float4*)(nodes + (size_t)bmu * D_DIM + c4);
    float4 o;
    o.x = zv.x + 0.1f * (nv.x - zv.x);
    o.y = zv.y + 0.1f * (nv.y - zv.y);
    o.z = zv.z + 0.1f * (nv.z - zv.z);
    o.w = zv.w + 0.1f * (nv.w - zv.w);
    *(float4*)(out_som + grow * (long long)D_DIM + c4) = o;
  }
}

// ---------------------------------------------------------------------------
extern "C" void kernel_launch(void* const* d_in, const int* in_sizes, int n_in,
                              void* d_out, int out_size, void* d_ws, size_t ws_size,
                              hipStream_t stream) {
  (void)in_sizes; (void)n_in; (void)out_size; (void)ws_size;
  const float* z     = (const float*)d_in[0];   // [64,1024,512] f32
  const float* nodes = (const float*)d_in[1];   // [32,32,512]   f32

  // workspace: nodes_hi (1 MB) | nodes_lo (1 MB) | n_sq (4 KB)  ~= 2.1 MB
  __bf16* nh = (__bf16*)d_ws;
  __bf16* nl = nh + (size_t)K_NODES * D_DIM;
  float* nsq = (float*)(nl + (size_t)K_NODES * D_DIM);

  float* out_som = (float*)d_out;                                   // 33.55M f32
  float* out_q   = out_som + (size_t)B_BATCH * T_SEQ * D_DIM;       // 67.1M f32
  int*   out_bmu = (int*)(out_q + (size_t)B_BATCH * T_SEQ * K_NODES); // 65536 i32

  som_prep_nodes<<<K_NODES, 256, 0, stream>>>(nodes, nh, nl, nsq);

  const int nblocks = (B_BATCH * T_SEQ) / ROWS;     // 2048
  const int lds_bytes = ROWS * D_DIM * 2 * (int)sizeof(__bf16);  // 64 KB
  som_main<<<nblocks, THREADS, lds_bytes, stream>>>(
      z, nodes, nh, nl, nsq, out_som, out_q, out_bmu);
}
